// MultiHeadAttention_87402584474092
// MI455X (gfx1250) — matrix operand order repaired and checked by reading, not compiled, not run on hardware
//
#include <hip/hip_runtime.h>

// ---------------- problem constants ----------------
#define DIM    768
#define HEADS  16
#define DK     48          // head dim
#define DKP    64          // head dim zero-padded to 2x WMMA K
#define BATCH  2
#define SEQ    2048
#define BS     (BATCH*SEQ) // 4096 total rows
#define WAVES  4           // waves per block (128 threads, wave32)
#define KTILE  32          // keys per attention iteration

typedef __attribute__((ext_vector_type(16))) _Float16 v16h;
typedef __attribute__((ext_vector_type(8)))  _Float16 v8h;
typedef __attribute__((ext_vector_type(4)))  _Float16 v4h;
typedef __attribute__((ext_vector_type(8)))  float    v8f;
typedef __attribute__((ext_vector_type(4)))  unsigned int u32x4;
typedef __attribute__((ext_vector_type(8)))  int      i32x8;
typedef __attribute__((ext_vector_type(4)))  int      i32x4;

#if defined(__HIP_DEVICE_COMPILE__) && __has_builtin(__builtin_amdgcn_tensor_load_to_lds)
#define USE_TDM 1
#else
#define USE_TDM 0
#endif

// ---------------- WMMA fragment helpers ----------------
// A-matrix 16x32 f16 (ISA 7.12.2): lane L (row = L&15, hi = L>>4) holds
// K = {8*hi..8*hi+7} in v0..v3 and K = {16+8*hi..23+8*hi} in v4..v7.
__device__ __forceinline__ v16h frag_a(const _Float16* rowptr, int hi) {
  union { v16h v; v8h h[2]; } u;
  u.h[0] = *(const v8h*)(rowptr + hi * 8);
  u.h[1] = *(const v8h*)(rowptr + 16 + hi * 8);
  return u.v;
}
// B-matrix 32x16 f16: lane L is column N = L&15; half hi holds
// K = {8*hi..8*hi+7} and {16+8*hi..23+8*hi}. rowptr = row N of B^T (row-major).
__device__ __forceinline__ v16h frag_b(const _Float16* rowptr, int hi) {
  union { v16h v; v8h h[2]; } u;
  u.h[0] = *(const v8h*)(rowptr + hi * 8);
  u.h[1] = *(const v8h*)(rowptr + 16 + hi * 8);
  return u.v;
}
__device__ __forceinline__ v8f wmma_f16(v16h a, v16h b, v8f c) {
  return __builtin_amdgcn_wmma_f32_16x16x32_f16(
      false, a, false, b, (short)0, c, false, false);
}

// ---------------- Tensor Data Mover (gfx1250 TDM) ----------------
#if USE_TDM
// LDS byte offset of a __shared__ object (addrspace(3) pointer value == offset)
__device__ __forceinline__ unsigned lds_off(const void* p) {
  return (unsigned)(unsigned long long)(const __attribute__((address_space(3))) char*)p;
}
// 2-D f16 tile DMA: global (row stride stride0 elems) -> LDS (packed rows).
// D# per ISA 8.3/8.4: count=1, type=2, data_size=1 (2 bytes), no pad/iterate.
__device__ __forceinline__ void tdm_load_2d(unsigned ldsoff, const void* gptr,
                                            unsigned tensor_d0, unsigned tensor_d1,
                                            unsigned tile_d0, unsigned tile_d1,
                                            unsigned long long stride0) {
  unsigned long long ga = (unsigned long long)gptr;
  u32x4 g0;
  g0[0] = 1u;                                   // count=1 (valid user D#)
  g0[1] = ldsoff;                               // lds_addr (bytes)
  g0[2] = (unsigned)(ga & 0xffffffffu);         // global_addr[31:0]
  g0[3] = (unsigned)((ga >> 32) & 0x1ffffffu) | (2u << 30);   // [56:32] | type=2
  i32x8 g1;
  g1[0] = (int)(1u << 16);                      // wg_mask=0, data_size=2B
  g1[1] = (int)((tensor_d0 & 0xffffu) << 16);                 // dim0[15:0]
  g1[2] = (int)(((tensor_d0 >> 16) & 0xffffu) | ((tensor_d1 & 0xffffu) << 16));
  g1[3] = (int)(((tensor_d1 >> 16) & 0xffffu) | ((tile_d0 & 0xffffu) << 16));
  g1[4] = (int)(tile_d1 & 0xffffu);             // tile_dim1, tile_dim2=0
  g1[5] = (int)(stride0 & 0xffffffffu);         // dim0_stride[31:0]
  g1[6] = (int)((stride0 >> 32) & 0xffffu);     // dim0_stride[47:32], dim1_stride=0
  g1[7] = 0;
  i32x4 z4 = {0, 0, 0, 0};
#if __clang_major__ >= 23
  i32x8 z8 = {0, 0, 0, 0, 0, 0, 0, 0};
  __builtin_amdgcn_tensor_load_to_lds(g0, g1, z4, z4, z8, 0);
#else
  __builtin_amdgcn_tensor_load_to_lds(g0, g1, z4, z4, 0);
#endif
}
#endif // USE_TDM

// ---------------- fp32 -> f16 conversion ----------------
__global__ void cvt_kernel(const float* __restrict__ src,
                           _Float16* __restrict__ dst, int n) {
  int i = (blockIdx.x * blockDim.x + threadIdx.x) * 4;
  if (i + 4 <= n) {
    const float4 f = *(const float4*)(src + i);
    v4h h;
    h.x = (_Float16)f.x; h.y = (_Float16)f.y;
    h.z = (_Float16)f.z; h.w = (_Float16)f.w;
    *(v4h*)(dst + i) = h;
  } else {
    for (; i < n; ++i) dst[i] = (_Float16)src[i];
  }
}

// ---------------- fused QKV projection ----------------
// y = x @ W^T + b.  Q scaled by 1/sqrt(dk), stored [B,H,S,DKP] (pad zeroed).
// K stored [B,H,S,DKP].  V stored transposed [B,H,DKP,S].
// Fragments software-pipelined so global loads overlap WMMA.
__global__ void __launch_bounds__(32 * WAVES)
qkv_kernel(const _Float16* __restrict__ xf,
           const _Float16* __restrict__ wq,
           const _Float16* __restrict__ wk,
           const _Float16* __restrict__ wv,
           const float* __restrict__ bq,
           const float* __restrict__ bk,
           const float* __restrict__ bv,
           _Float16* __restrict__ qf,
           _Float16* __restrict__ kf,
           _Float16* __restrict__ vt) {
  const int wave = threadIdx.x >> 5;
  const int lane = threadIdx.x & 31;
  const int r = lane & 15, hi = lane >> 4;
  const int m0 = blockIdx.x * 16;
  const int n0 = (blockIdx.y * WAVES + wave) * 16;

  const _Float16* xrow = xf + (size_t)(m0 + r) * DIM;
  const _Float16* wqr  = wq + (size_t)(n0 + r) * DIM;
  const _Float16* wkr  = wk + (size_t)(n0 + r) * DIM;
  const _Float16* wvr  = wv + (size_t)(n0 + r) * DIM;

  v8f cq = {}, ck = {}, cv = {};
  v16h a  = frag_a(xrow, hi);
  v16h fq = frag_b(wqr, hi);
  v16h fk = frag_b(wkr, hi);
  v16h fv = frag_b(wvr, hi);
  for (int k = 0; k < DIM; k += 32) {
    v16h a2 = {}, fq2 = {}, fk2 = {}, fv2 = {};
    if (k + 32 < DIM) {                 // prefetch next fragments
      a2  = frag_a(xrow + k + 32, hi);
      fq2 = frag_b(wqr + k + 32, hi);
      fk2 = frag_b(wkr + k + 32, hi);
      fv2 = frag_b(wvr + k + 32, hi);
    }
    cq = wmma_f16(a, fq, cq);
    ck = wmma_f16(a, fk, ck);
    cv = wmma_f16(a, fv, cv);
    a = a2; fq = fq2; fk = fk2; fv = fv2;
  }

  const int h  = n0 / DK;          // 48 = 3*16: each 16-col tile within one head
  const int c0 = n0 - h * DK;      // 0, 16, 32
  const float qscale = 0.14433756729740643f;  // 1/sqrt(48)
#pragma unroll
  for (int i = 0; i < 8; i++) {
    const int mrow = m0 + i + 8 * hi;
    const int b = mrow >> 11;
    const int s = mrow & (SEQ - 1);
    const int n = n0 + r;
    const size_t bh = (size_t)b * HEADS + h;
    const size_t qkoff = (bh * SEQ + s) * DKP + c0 + r;
    qf[qkoff] = (_Float16)((cq[i] + bq[n]) * qscale);
    kf[qkoff] = (_Float16)(ck[i] + bk[n]);
    vt[(bh * DKP + (c0 + r)) * SEQ + s] = (_Float16)(cv[i] + bv[n]);
    if (c0 == 32) {                           // zero dk pad 48..63
      qf[qkoff + 16] = (_Float16)0.f;
      kf[qkoff + 16] = (_Float16)0.f;
    }
  }
}

// ---------------- flash attention ----------------
// One wave per 16-query tile, 32 keys/iteration.  K/V tiles staged to LDS
// via TDM (double buffered, s_wait_tensorcnt), P transposed through LDS.
__global__ void __launch_bounds__(32 * WAVES)
attn_kernel(const _Float16* __restrict__ qf,
            const _Float16* __restrict__ kf,
            const _Float16* __restrict__ vt,
            _Float16* __restrict__ ctxf) {
#if USE_TDM
  __shared__ _Float16 Kbuf[WAVES][2][KTILE * DKP];  // 32 KB
  __shared__ _Float16 Vbuf[WAVES][2][DK * KTILE];   // 24 KB
#endif
  __shared__ _Float16 Pbuf[WAVES][16 * KTILE];      // 4 KB

  const int wave = __builtin_amdgcn_readfirstlane((int)(threadIdx.x >> 5));
  const int lane = threadIdx.x & 31;
  const int r = lane & 15, hi = lane >> 4;
  const int qtile = blockIdx.x * WAVES + wave;
  const int bh = qtile >> 7;                  // SEQ/16 = 128 qtiles per (b,h)
  const int qt = qtile & 127;

  const _Float16* qbase = qf + ((size_t)bh * SEQ + (size_t)qt * 16) * DKP;
  const _Float16* kbase = kf + (size_t)bh * SEQ * DKP;
  const _Float16* vbase = vt + (size_t)bh * DKP * SEQ;

  const _Float16* qrow = qbase + (size_t)r * DKP;
  const v16h qa0 = frag_a(qrow, hi);
  const v16h qa1 = frag_a(qrow + 32, hi);

  v8f acc0 = {}, acc1 = {}, acc2 = {};
  float mr[8], lr[8];
#pragma unroll
  for (int i = 0; i < 8; i++) { mr[i] = -3.0e38f; lr[i] = 0.f; }

  _Float16* P = &Pbuf[wave][0];

#if USE_TDM
  // prologue: DMA first K (32x64) and V (48x32) tiles into buffer 0
  tdm_load_2d(lds_off(&Kbuf[wave][0][0]), kbase, DKP, KTILE, DKP, KTILE, DKP);
  tdm_load_2d(lds_off(&Vbuf[wave][0][0]), vbase, KTILE, DK, KTILE, DK, SEQ);
#endif

  for (int j = 0; j < SEQ; j += KTILE) {
#if USE_TDM
    const int cur = (j >> 5) & 1;
    if (j + KTILE < SEQ) {                     // DMA next tiles, then wait cur
      tdm_load_2d(lds_off(&Kbuf[wave][cur ^ 1][0]),
                  kbase + (size_t)(j + KTILE) * DKP, DKP, KTILE, DKP, KTILE, DKP);
      tdm_load_2d(lds_off(&Vbuf[wave][cur ^ 1][0]),
                  vbase + (j + KTILE), KTILE, DK, KTILE, DK, SEQ);
      __builtin_amdgcn_s_wait_tensorcnt(2);    // in-order: current pair landed
    } else {
      __builtin_amdgcn_s_wait_tensorcnt(0);
    }
    const _Float16* Kc = &Kbuf[wave][cur][0];  // 32 rows x DKP, packed
    const _Float16* Vc = &Vbuf[wave][cur][0];  // 48 rows x KTILE, packed
    const _Float16* kr0 = Kc + (size_t)r * DKP;
    const _Float16* kr1 = Kc + (size_t)(16 + r) * DKP;
    const _Float16* vr0 = Vc + (size_t)(0  + r) * KTILE;
    const _Float16* vr1 = Vc + (size_t)(16 + r) * KTILE;
    const _Float16* vr2 = Vc + (size_t)(32 + r) * KTILE;
#else
    const _Float16* kr0 = kbase + (size_t)(j + r) * DKP;
    const _Float16* kr1 = kbase + (size_t)(j + 16 + r) * DKP;
    const _Float16* vr0 = vbase + (size_t)(0  + r) * SEQ + j;
    const _Float16* vr1 = vbase + (size_t)(16 + r) * SEQ + j;
    const _Float16* vr2 = vbase + (size_t)(32 + r) * SEQ + j;
#endif
    // scores for keys [j, j+32): two 16x16 C-tiles, K=64 in two WMMA steps
    v8f s0 = {}, s1 = {};
    s0 = wmma_f16(qa0, frag_b(kr0, hi), s0);
    s0 = wmma_f16(qa1, frag_b(kr0 + 32, hi), s0);
    s1 = wmma_f16(qa0, frag_b(kr1, hi), s1);
    s1 = wmma_f16(qa1, frag_b(kr1 + 32, hi), s1);

    // online softmax: C layout => VGPR i is row (i + 8*hi), column = lane&15
#pragma unroll
    for (int i = 0; i < 8; i++) {
      float a = s0[i], b = s1[i];
      float mx = fmaxf(a, b);
      mx = fmaxf(mx, __shfl_xor(mx, 1, 32));
      mx = fmaxf(mx, __shfl_xor(mx, 2, 32));
      mx = fmaxf(mx, __shfl_xor(mx, 4, 32));
      mx = fmaxf(mx, __shfl_xor(mx, 8, 32));
      const float mnew = fmaxf(mr[i], mx);
      const float corr = __expf(mr[i] - mnew);
      const float p0 = __expf(a - mnew);
      const float p1 = __expf(b - mnew);
      float ps = p0 + p1;
      ps += __shfl_xor(ps, 1, 32);
      ps += __shfl_xor(ps, 2, 32);
      ps += __shfl_xor(ps, 4, 32);
      ps += __shfl_xor(ps, 8, 32);
      lr[i] = lr[i] * corr + ps;
      mr[i] = mnew;
      acc0[i] *= corr; acc1[i] *= corr; acc2[i] *= corr;
      const int prow = i + 8 * hi;
      P[prow * KTILE + r]      = (_Float16)p0;
      P[prow * KTILE + 16 + r] = (_Float16)p1;
    }
    asm volatile("s_wait_dscnt 0" ::: "memory");  // P stores -> P loads

    const v16h pa = frag_a(P + (size_t)r * KTILE, hi);
    acc0 = wmma_f16(pa, frag_b(vr0, hi), acc0);
    acc1 = wmma_f16(pa, frag_b(vr1, hi), acc1);
    acc2 = wmma_f16(pa, frag_b(vr2, hi), acc2);
  }

  const int b = bh >> 4;
  const int h = bh & 15;
#pragma unroll
  for (int i = 0; i < 8; i++) {
    const int s = qt * 16 + i + 8 * hi;
    const float inv = 1.0f / lr[i];
    const size_t off = ((size_t)(b * SEQ + s)) * DIM + h * DK;
    ctxf[off + r]      = (_Float16)(acc0[i] * inv);
    ctxf[off + 16 + r] = (_Float16)(acc1[i] * inv);
    ctxf[off + 32 + r] = (_Float16)(acc2[i] * inv);
  }
}

// ---------------- output projection ----------------
__global__ void __launch_bounds__(32 * WAVES)
out_proj_kernel(const _Float16* __restrict__ cf,
                const _Float16* __restrict__ wo,
                const float* __restrict__ bo,
                float* __restrict__ out) {
  const int wave = threadIdx.x >> 5;
  const int lane = threadIdx.x & 31;
  const int r = lane & 15, hi = lane >> 4;
  const int m0 = blockIdx.x * 16;
  const int n0 = (blockIdx.y * WAVES + wave) * 16;

  const _Float16* crow = cf + (size_t)(m0 + r) * DIM;
  const _Float16* wor  = wo + (size_t)(n0 + r) * DIM;
  v8f acc = {};
  v16h a = frag_a(crow, hi);
  v16h w = frag_b(wor, hi);
  for (int k = 0; k < DIM; k += 32) {
    v16h a2 = {}, w2 = {};
    if (k + 32 < DIM) {
      a2 = frag_a(crow + k + 32, hi);
      w2 = frag_b(wor + k + 32, hi);
    }
    acc = wmma_f16(a, w, acc);
    a = a2; w = w2;
  }
#pragma unroll
  for (int i = 0; i < 8; i++) {
    const int mrow = m0 + i + 8 * hi;
    out[(size_t)mrow * DIM + n0 + r] = acc[i] + bo[n0 + r];
  }
}

// ---------------- host launcher ----------------
extern "C" void kernel_launch(void* const* d_in, const int* in_sizes, int n_in,
                              void* d_out, int out_size, void* d_ws, size_t ws_size,
                              hipStream_t stream) {
  (void)in_sizes; (void)n_in; (void)out_size; (void)ws_size;
  const float* x  = (const float*)d_in[0];
  const float* Wq = (const float*)d_in[1];
  const float* bq = (const float*)d_in[2];
  const float* Wk = (const float*)d_in[3];
  const float* bk = (const float*)d_in[4];
  const float* Wv = (const float*)d_in[5];
  const float* bv = (const float*)d_in[6];
  const float* Wo = (const float*)d_in[7];
  const float* bo = (const float*)d_in[8];
  float* out = (float*)d_out;

  char* ws = (char*)d_ws;
  _Float16* xf16  = (_Float16*)ws; ws += (size_t)BS * DIM * 2;
  _Float16* wq16  = (_Float16*)ws; ws += (size_t)DIM * DIM * 2;
  _Float16* wk16  = (_Float16*)ws; ws += (size_t)DIM * DIM * 2;
  _Float16* wv16  = (_Float16*)ws; ws += (size_t)DIM * DIM * 2;
  _Float16* wo16  = (_Float16*)ws; ws += (size_t)DIM * DIM * 2;
  _Float16* qf16  = (_Float16*)ws; ws += (size_t)BATCH * HEADS * SEQ * DKP * 2;
  _Float16* kf16  = (_Float16*)ws; ws += (size_t)BATCH * HEADS * SEQ * DKP * 2;
  _Float16* vt16  = (_Float16*)ws; ws += (size_t)BATCH * HEADS * DKP * SEQ * 2;
  _Float16* ctx16 = (_Float16*)ws; ws += (size_t)BS * DIM * 2;

  const int nx = BS * DIM;
  const int nw = DIM * DIM;
  cvt_kernel<<<(nx / 4 + 255) / 256, 256, 0, stream>>>(x,  xf16, nx);
  cvt_kernel<<<(nw / 4 + 255) / 256, 256, 0, stream>>>(Wq, wq16, nw);
  cvt_kernel<<<(nw / 4 + 255) / 256, 256, 0, stream>>>(Wk, wk16, nw);
  cvt_kernel<<<(nw / 4 + 255) / 256, 256, 0, stream>>>(Wv, wv16, nw);
  cvt_kernel<<<(nw / 4 + 255) / 256, 256, 0, stream>>>(Wo, wo16, nw);

  dim3 gproj(BS / 16, (DIM / 16) / WAVES);            // (256, 12)
  qkv_kernel<<<gproj, 32 * WAVES, 0, stream>>>(
      xf16, wq16, wk16, wv16, bq, bk, bv, qf16, kf16, vt16);

  const int nqtiles = BATCH * HEADS * (SEQ / 16);     // 4096
  attn_kernel<<<nqtiles / WAVES, 32 * WAVES, 0, stream>>>(
      qf16, kf16, vt16, ctx16);

  out_proj_kernel<<<gproj, 32 * WAVES, 0, stream>>>(ctx16, wo16, bo, out);
}
